// GConvLSTMCell_34479997453150
// MI455X (gfx1250) — matrix operand order repaired and checked by reading, compiled, not run on hardware
//
#include <hip/hip_runtime.h>
#include <hip/hip_bf16.h>
#include <math.h>

// ---------------------------------------------------------------------------
// GConvLSTM cell for MI455X (gfx1250, wave32, WMMA).
//   Stage 1: swizzle [x||h] and [Wi||Wf||Wo||Wg] into bf16 WMMA fragment layout
//   Stage 2: xt = xh @ W via v_wmma_f32_16x16x32_bf16 (32x64 tile per wave)
//   Stage 3: agg = bias + selfloop + scatter(norm * xt[src])  (L2-resident atomics)
//   Stage 4: LSTM gates -> [h_new || c_new]
// ---------------------------------------------------------------------------

typedef __attribute__((ext_vector_type(16))) __bf16 v16bf;
typedef __attribute__((ext_vector_type(8)))  float  v8f;

__device__ __forceinline__ __bf16 f2bf(float f) {
  // round-to-nearest-even f32 -> bf16 (inputs are finite)
  unsigned u = __builtin_bit_cast(unsigned, f);
  unsigned r = u + 0x7FFFu + ((u >> 16) & 1u);
  unsigned short s = (unsigned short)(r >> 16);
  return __builtin_bit_cast(__bf16, s);
}

__device__ __forceinline__ float sigm(float x) { return 1.0f / (1.0f + expf(-x)); }

// ---- 1) degree / normalization -------------------------------------------
__global__ void k_init_deg(float* deg, int N) {
  int n = blockIdx.x * blockDim.x + threadIdx.x;
  if (n < N) deg[n] = 1.0f;  // self-loop weight
}

__global__ void k_deg_accum(const int* __restrict__ ei, const float* __restrict__ ew,
                            float* deg, int E) {
  int e = blockIdx.x * blockDim.x + threadIdx.x;
  if (e < E) atomicAdd(&deg[ei[E + e]], ew[e]);  // segment_sum over dst
}

__global__ void k_dinv(const float* __restrict__ deg, float* dinv, int N) {
  int n = blockIdx.x * blockDim.x + threadIdx.x;
  if (n < N) {
    float d = deg[n];
    dinv[n] = (d > 0.0f) ? rsqrtf(d) : 0.0f;
  }
}

__global__ void k_norm(const int* __restrict__ ei, const float* __restrict__ ew,
                       const float* __restrict__ dinv, float* normv, int E) {
  int e = blockIdx.x * blockDim.x + threadIdx.x;
  if (e < E) normv[e] = dinv[ei[e]] * ew[e] * dinv[ei[E + e]];
}

// ---- 2a) A-operand swizzle: [x||h] f32 -> bf16 in WMMA A-fragment layout --
// One thread per (mTile, kChunk, lane); writes that lane's 16 bf16 (32 B).
// A frag k order used here (contiguous, not AMD table order): lane<16: M=lane, elems 0..15 -> K=k0..k0+15;
//                        lane>=16: K=k0+16..k0+31 (matches B fragments).
__global__ void k_xh_frag(const float* __restrict__ x, const float* __restrict__ h,
                          __bf16* __restrict__ xhb, int Cdim, int dK, int total32) {
  int t = blockIdx.x * blockDim.x + threadIdx.x;
  if (t >= total32) return;
  int lane = t & 31;
  int tq = t >> 5;                 // tile*Q + q
  int Q = dK >> 5;
  int tile = tq / Q, q = tq - tile * Q;
  int r = lane & 15, half = lane >> 4;
  int row = tile * 16 + r;
  int Hdim = dK - Cdim;
  const float* __restrict__ xrow = x + (size_t)row * Cdim;
  const float* __restrict__ hrow = h + (size_t)row * Hdim;
  int k0 = q * 32 + half * 16;     // 8-aligned; never straddles x|h seam (Cdim%8==0)
  int k1 = k0 + 8;
  const float* p0 = (k0 < Cdim) ? (xrow + k0) : (hrow + (k0 - Cdim));
  const float* p1 = (k1 < Cdim) ? (xrow + k1) : (hrow + (k1 - Cdim));
  float4 v0 = *(const float4*)p0;
  float4 v1 = *(const float4*)(p0 + 4);
  float4 v2 = *(const float4*)p1;
  float4 v3 = *(const float4*)(p1 + 4);
  v16bf a;
  a[0]  = f2bf(v0.x); a[1]  = f2bf(v0.y); a[2]  = f2bf(v0.z); a[3]  = f2bf(v0.w);
  a[4]  = f2bf(v1.x); a[5]  = f2bf(v1.y); a[6]  = f2bf(v1.z); a[7]  = f2bf(v1.w);
  a[8]  = f2bf(v2.x); a[9]  = f2bf(v2.y); a[10] = f2bf(v2.z); a[11] = f2bf(v2.w);
  a[12] = f2bf(v3.x); a[13] = f2bf(v3.y); a[14] = f2bf(v3.z); a[15] = f2bf(v3.w);
  *(v16bf*)(xhb + (size_t)t * 16) = a;
}

// ---- 2b) B-operand swizzle: W -> bf16 WMMA B-fragment layout --------------
// B frag k order used here (contiguous, same as A): lane<16: N=lane, K=k0..k0+15; lane>=16: N=lane-16, K=k0+16..31.
__global__ void k_w_frag(const float* __restrict__ Wi, const float* __restrict__ Wf,
                         const float* __restrict__ Wo, const float* __restrict__ Wg,
                         __bf16* __restrict__ Wtf, int dK, int H, int total32) {
  int t = blockIdx.x * blockDim.x + threadIdx.x;
  if (t >= total32) return;
  int lane = t & 31;
  int cq = t >> 5;                 // colBlock*Q + q
  int Q = dK >> 5;
  int cb = cq / Q, q = cq - cb * Q;
  int n = lane & 15, half = lane >> 4;
  int col = cb * 16 + n;
  int gate = col / H, j = col - gate * H;
  const float* W = (gate == 0) ? Wi : (gate == 1) ? Wf : (gate == 2) ? Wo : Wg;
  int k0 = q * 32 + half * 16;
  v16bf b;
#pragma unroll
  for (int e = 0; e < 16; ++e) b[e] = f2bf(W[(size_t)(k0 + e) * H + j]);
  *(v16bf*)(Wtf + (size_t)t * 16) = b;
}

// ---- 3) WMMA GEMM: xt = xh @ W ; 32 rows x 64 cols per wave ---------------
// 1 block = 8 waves; wave w = 64-col strip w; block covers rows [32*blk, 32*blk+32).
// Inner loop: 2 A frags + 4 B frags (six 32B coalesced loads) -> 8 WMMAs.
__global__ __launch_bounds__(256) void k_gemm_wmma(
    const __bf16* __restrict__ xhb, const __bf16* __restrict__ Wtf,
    float* __restrict__ xt, int BN, int dK, int H4) {
  const int wave = threadIdx.x >> 5;  // 0..7 == column strip
  const int lane = threadIdx.x & 31;
  const int r15 = lane & 15;
  const int half = lane >> 4;
  const int Q = dK >> 5;
  const int mt0 = blockIdx.x * 2;
  const int mt1 = mt0 + 1;
  if (mt1 * 16 >= BN + 16) return;  // uniform guard (BN % 32 == 0)

  const __bf16* aBase0 = xhb + ((size_t)mt0 * Q * 32 + lane) * 16;
  const __bf16* aBase1 = xhb + ((size_t)mt1 * Q * 32 + lane) * 16;
  const __bf16* bBase0 = Wtf + ((size_t)(wave * 4 + 0) * Q * 32 + lane) * 16;
  const __bf16* bBase1 = Wtf + ((size_t)(wave * 4 + 1) * Q * 32 + lane) * 16;
  const __bf16* bBase2 = Wtf + ((size_t)(wave * 4 + 2) * Q * 32 + lane) * 16;
  const __bf16* bBase3 = Wtf + ((size_t)(wave * 4 + 3) * Q * 32 + lane) * 16;

  v8f acc[8] = {};
#pragma unroll 4
  for (int q = 0; q < Q; ++q) {
    const size_t qo = (size_t)q * 512;  // 32 lanes * 16 bf16 per chunk
    v16bf a0 = *(const v16bf*)(aBase0 + qo);
    v16bf a1 = *(const v16bf*)(aBase1 + qo);
    if (q + 1 < Q) {  // hint next A slabs into cache (global_prefetch_b8)
      __builtin_prefetch(aBase0 + qo + 512, 0, 1);
      __builtin_prefetch(aBase1 + qo + 512, 0, 1);
    }
    v16bf b0 = *(const v16bf*)(bBase0 + qo);
    v16bf b1 = *(const v16bf*)(bBase1 + qo);
    v16bf b2 = *(const v16bf*)(bBase2 + qo);
    v16bf b3 = *(const v16bf*)(bBase3 + qo);
    acc[0] = __builtin_amdgcn_wmma_f32_16x16x32_bf16(false, a0, false, b0, (short)0, acc[0], false, false);
    acc[1] = __builtin_amdgcn_wmma_f32_16x16x32_bf16(false, a0, false, b1, (short)0, acc[1], false, false);
    acc[2] = __builtin_amdgcn_wmma_f32_16x16x32_bf16(false, a0, false, b2, (short)0, acc[2], false, false);
    acc[3] = __builtin_amdgcn_wmma_f32_16x16x32_bf16(false, a0, false, b3, (short)0, acc[3], false, false);
    acc[4] = __builtin_amdgcn_wmma_f32_16x16x32_bf16(false, a1, false, b0, (short)0, acc[4], false, false);
    acc[5] = __builtin_amdgcn_wmma_f32_16x16x32_bf16(false, a1, false, b1, (short)0, acc[5], false, false);
    acc[6] = __builtin_amdgcn_wmma_f32_16x16x32_bf16(false, a1, false, b2, (short)0, acc[6], false, false);
    acc[7] = __builtin_amdgcn_wmma_f32_16x16x32_bf16(false, a1, false, b3, (short)0, acc[7], false, false);
  }

  // ---- D layout (f32 16x16, 8 VGPRs): VGPR r -> M = r + 8*half, N = lane&15
  const int colW = wave * 64 + r15;
#pragma unroll
  for (int j = 0; j < 4; ++j) {
#pragma unroll
    for (int r = 0; r < 8; ++r) {
      int m0 = mt0 * 16 + r + 8 * half;
      int m1 = mt1 * 16 + r + 8 * half;
      xt[(size_t)m0 * H4 + colW + j * 16] = acc[j][r];
      xt[(size_t)m1 * H4 + colW + j * 16] = acc[4 + j][r];
    }
  }
}

// ---- 4) agg init: bias + self-loop term (norm = dinv[n]^2) ---------------
__global__ void k_agg_init(const float* __restrict__ xt, const float* __restrict__ dinv,
                           const float* __restrict__ bi, const float* __restrict__ bf_,
                           const float* __restrict__ bo, const float* __restrict__ bg,
                           float* __restrict__ agg, int N, int H, int H4, long long total) {
  long long idx = (long long)blockIdx.x * blockDim.x + threadIdx.x;
  if (idx >= total) return;
  int f = (int)(idx % H4);
  int n = (int)((idx / H4) % N);
  int gate = f / H, j = f - gate * H;
  float bias = (gate == 0) ? bi[j] : (gate == 1) ? bf_[j] : (gate == 2) ? bo[j] : bg[j];
  float dv = dinv[n];
  agg[idx] = bias + xt[idx] * dv * dv;
}

// ---- 5) edge scatter: agg[b,dst,:] += norm[e] * xt[b,src,:] --------------
// blockDim.x == H4/4; float4 read, 4x global_atomic_add_f32 (agg stays in L2)
__global__ void k_scatter(const int* __restrict__ ei, const float* __restrict__ normv,
                          const float* __restrict__ xt, float* __restrict__ agg,
                          int Nn, int E, int H4) {
  int e = blockIdx.x;
  int b = blockIdx.y;
  int f = threadIdx.x * 4;
  int src = ei[e];
  int dst = ei[E + e];
  float w = normv[e];
  const float4 v = *(const float4*)(xt + ((size_t)b * Nn + src) * H4 + f);
  float* ap = agg + ((size_t)b * Nn + dst) * H4 + f;
  atomicAdd(ap + 0, v.x * w);
  atomicAdd(ap + 1, v.y * w);
  atomicAdd(ap + 2, v.z * w);
  atomicAdd(ap + 3, v.w * w);
}

// ---- 6) LSTM gates -> d_out = [h_new || c_new] ---------------------------
__global__ void k_gates(const float* __restrict__ agg, const float* __restrict__ c,
                        float* __restrict__ out, int H, int H4, long long BNH) {
  long long idx = (long long)blockIdx.x * blockDim.x + threadIdx.x;
  if (idx >= BNH) return;
  long long row = idx / H;
  int hd = (int)(idx % H);
  const float* a = agg + row * (long long)H4;
  float ig = sigm(a[hd]);
  float fg = sigm(a[H + hd]);
  float og = sigm(a[2 * H + hd]);
  float gg = tanhf(a[3 * H + hd]);
  float cn = fg * c[idx] + ig * gg;
  float hn = og * tanhf(cn);
  out[idx] = hn;
  out[BNH + idx] = cn;
}

// ---------------------------------------------------------------------------
extern "C" void kernel_launch(void* const* d_in, const int* in_sizes, int n_in,
                              void* d_out, int out_size, void* d_ws, size_t ws_size,
                              hipStream_t stream) {
  const float* x  = (const float*)d_in[0];
  const float* h  = (const float*)d_in[1];
  const float* c  = (const float*)d_in[2];
  const int*   ei = (const int*)d_in[3];
  const float* ew = (const float*)d_in[4];
  const float* Wi = (const float*)d_in[5];
  const float* bi = (const float*)d_in[6];
  const float* Wf = (const float*)d_in[7];
  const float* bf = (const float*)d_in[8];
  const float* Wo = (const float*)d_in[9];
  const float* bo = (const float*)d_in[10];
  const float* Wg = (const float*)d_in[11];
  const float* bg = (const float*)d_in[12];

  const int H    = in_sizes[6];            // 128
  const int dK   = in_sizes[5] / H;        // C+H = 256
  const int Cdim = dK - H;                 // 128
  const int E    = in_sizes[4];            // 160000
  const int BN   = in_sizes[1] / H;        // B*N = 40000
  const int B    = 4;                      // problem constant
  const int N    = BN / B;                 // 10000
  const int H4   = 4 * H;                  // 512

  // workspace carve-up (32B aligned sections)
  float* ws    = (float*)d_ws;
  float* deg   = ws;
  float* dinv  = deg + N;
  float* normv = dinv + N;
  size_t off = (size_t)2 * N + E;
  off = (off + 7) & ~(size_t)7;
  __bf16* xhb = (__bf16*)(ws + off);                // A fragments: [BN/16][Q][32][16] bf16
  size_t xhbFloats = ((size_t)BN * dK + 1) / 2;
  off += (xhbFloats + 7) & ~(size_t)7;
  __bf16* Wtf = (__bf16*)(ws + off);                // B fragments: [H4/16][Q][32][16] bf16
  size_t wtFloats = ((size_t)dK * H4 + 1) / 2;
  off += (wtFloats + 7) & ~(size_t)7;
  float* xt  = ws + off;                            // [BN][H4]
  off += (size_t)BN * H4;
  float* agg = ws + off;                            // [BN][H4]

  const int T = 256;
  hipLaunchKernelGGL(k_init_deg,  dim3((N + T - 1) / T), dim3(T), 0, stream, deg, N);
  hipLaunchKernelGGL(k_deg_accum, dim3((E + T - 1) / T), dim3(T), 0, stream, ei, ew, deg, E);
  hipLaunchKernelGGL(k_dinv,      dim3((N + T - 1) / T), dim3(T), 0, stream, deg, dinv, N);
  hipLaunchKernelGGL(k_norm,      dim3((E + T - 1) / T), dim3(T), 0, stream, ei, ew, dinv, normv, E);

  const int Q = dK / 32;
  int aTotal = (BN / 16) * Q * 32;
  hipLaunchKernelGGL(k_xh_frag, dim3((aTotal + T - 1) / T), dim3(T), 0, stream,
                     x, h, xhb, Cdim, dK, aTotal);
  int bTotal = (H4 / 16) * Q * 32;
  hipLaunchKernelGGL(k_w_frag, dim3((bTotal + T - 1) / T), dim3(T), 0, stream,
                     Wi, Wf, Wo, Wg, Wtf, dK, H, bTotal);

  // GEMM: one block (8 waves) per 32-row tile; wave = 64-col strip (H4==512)
  hipLaunchKernelGGL(k_gemm_wmma, dim3(BN / 32), dim3(256), 0, stream,
                     xhb, Wtf, xt, BN, dK, H4);

  long long aggTotal = (long long)BN * H4;
  hipLaunchKernelGGL(k_agg_init, dim3((unsigned)((aggTotal + T - 1) / T)), dim3(T), 0, stream,
                     xt, dinv, bi, bf, bo, bg, agg, N, H, H4, aggTotal);

  hipLaunchKernelGGL(k_scatter, dim3(E, B), dim3(H4 / 4), 0, stream,
                     ei, normv, xt, agg, N, E, H4);

  long long BNH = (long long)BN * H;
  hipLaunchKernelGGL(k_gates, dim3((unsigned)((BNH + T - 1) / T)), dim3(T), 0, stream,
                     agg, c, (float*)d_out, H, H4, BNH);
}
